// GPT_77403900608944
// MI455X (gfx1250) — matrix operand order repaired and checked by reading, not compiled, not run on hardware
//
#include <hip/hip_runtime.h>
#include <hip/hip_bf16.h>

// ---------------------------------------------------------------------------
// Problem constants (from reference): B=1, T=2048, NE=1024, NH=16, NKV=4,
// HD=64, GC=32, window=1024, EPS=1.1920929e-07
// ---------------------------------------------------------------------------
#define T_    2048
#define NE_   1024
#define NH_   16
#define NKV_  4
#define HD_   64
#define GC_   32
#define WIN_  1024
#define EPS_  1.1920929e-07f

typedef __attribute__((ext_vector_type(16))) __bf16 v16bf;
typedef __attribute__((ext_vector_type(8)))  float  v8f;

union Frag {
    v16bf v;
    unsigned short u[16];
    uint4 q[2];
};

__device__ __forceinline__ unsigned short f2bf(float f) {
    unsigned int u = __float_as_uint(f);
    unsigned int r = (u + 0x7FFFu + ((u >> 16) & 1u)) >> 16;
    return (unsigned short)r;
}

__device__ __forceinline__ v8f zero8() {
    v8f z;
#pragma unroll
    for (int i = 0; i < 8; ++i) z[i] = 0.0f;
    return z;
}

// ---------------------------------------------------------------------------
// f32 -> bf16 convert (row-major copy)
// ---------------------------------------------------------------------------
__global__ void __launch_bounds__(256) cvt_bf16(const float* __restrict__ in,
                                                unsigned short* __restrict__ out,
                                                int n) {
    int i = blockIdx.x * 256 + threadIdx.x;
    if (i < n) out[i] = f2bf(in[i]);
}

// ---------------------------------------------------------------------------
// f32 (K x N, row-major) -> bf16 TRANSPOSED (N x K, row-major), tiled via LDS.
// Makes WMMA B-fragments contiguous 16B loads (lane -> column n, K along mem).
// Grid: (N/32, K/32), block (32, 8).
// ---------------------------------------------------------------------------
__global__ void __launch_bounds__(256) cvt_bf16_transpose(
    const float* __restrict__ in, unsigned short* __restrict__ out,
    int K, int N)
{
    __shared__ float tile[32][33];
    int n0 = blockIdx.x * 32;
    int k0 = blockIdx.y * 32;
    int tx = threadIdx.x;
#pragma unroll
    for (int i = threadIdx.y; i < 32; i += 8)
        tile[i][tx] = in[(size_t)(k0 + i) * N + n0 + tx];   // [k_local][n_local]
    __syncthreads();
#pragma unroll
    for (int i = threadIdx.y; i < 32; i += 8)
        out[(size_t)(n0 + i) * K + k0 + tx] = f2bf(tile[tx][i]);
}

// ---------------------------------------------------------------------------
// Fragment loader for the LDS-free GEMM: A fragment (two 16B loads) + four
// B fragments from transposed weights (two 16B loads each).
// ---------------------------------------------------------------------------
__device__ __forceinline__ void load_ab(const unsigned short* __restrict__ ap,
                                        const unsigned short* __restrict__ bp0,
                                        size_t K, int k0,
                                        Frag& a, Frag (&b)[4]) {
    a.q[0] = *(const uint4*)(ap + k0);
    a.q[1] = *(const uint4*)(ap + k0 + 16);
    __builtin_prefetch(ap + k0 + 32, 0, 1);
#pragma unroll
    for (int nt = 0; nt < 4; ++nt) {
        const unsigned short* bp = bp0 + (size_t)nt * 16 * K + k0;
        b[nt].q[0] = *(const uint4*)(bp);
        b[nt].q[1] = *(const uint4*)(bp + 16);
        __builtin_prefetch(bp + 32, 0, 1);
    }
}

// ---------------------------------------------------------------------------
// bf16 WMMA GEMM, LDS-free, 2-deep software pipelined:
//   C[M,N] = A[M,K] @ Bt[N,K]^T     (K % 64 == 0)
// A row-major (M,K) bf16; Bt row-major (N,K) bf16 (pre-transposed weights);
// C f32 row-major (M,N). Block = 128 threads (4 waves), tile 64(M) x 64(N);
// wave w owns rows [tileRow+16w, +16) x all 64 cols (4 accumulators).
// While WMMAs consume one fragment buffer, the other buffer's loads are in
// flight, so waits are partial instead of full drains.
// ---------------------------------------------------------------------------
__global__ void __launch_bounds__(128) gemm_bf16_wmma(
    const unsigned short* __restrict__ A,
    const unsigned short* __restrict__ Bt,
    float* __restrict__ C, int M, int N, int K)
{
    const int lane = threadIdx.x & 31;
    const int w    = threadIdx.x >> 5;
    const int tileRow = blockIdx.y * 64;
    const int tileCol = blockIdx.x * 64;
    const int half = (lane >> 4) & 1;   // which 16-lane half of the wave
    const int ln   = lane & 15;
    const int m    = tileRow + w * 16 + ln;

    v8f acc[4];
#pragma unroll
    for (int i = 0; i < 4; ++i) acc[i] = zero8();

    const unsigned short* ap  = A + (size_t)m * K + half * 8;
    const unsigned short* bp0 = Bt + (size_t)(tileCol + ln) * K + half * 8;

    Frag a0, a1;
    Frag b0[4], b1[4];
    load_ab(ap, bp0, K, 0, a0, b0);

    for (int k0 = 0; k0 < K; k0 += 64) {
        // stage 1 loads (k0+32) overlap with stage 0 WMMAs
        load_ab(ap, bp0, K, k0 + 32, a1, b1);
#pragma unroll
        for (int nt = 0; nt < 4; ++nt)
            acc[nt] = __builtin_amdgcn_wmma_f32_16x16x32_bf16(
                false, a0.v, false, b0[nt].v, (short)0, acc[nt], false, false);

        // stage 0 loads (k0+64) overlap with stage 1 WMMAs
        if (k0 + 64 < K)
            load_ab(ap, bp0, K, k0 + 64, a0, b0);
#pragma unroll
        for (int nt = 0; nt < 4; ++nt)
            acc[nt] = __builtin_amdgcn_wmma_f32_16x16x32_bf16(
                false, a1.v, false, b1[nt].v, (short)0, acc[nt], false, false);
    }

#pragma unroll
    for (int nt = 0; nt < 4; ++nt) {
#pragma unroll
        for (int r = 0; r < 8; ++r) {
            int row = tileRow + w * 16 + r + half * 8;
            int col = tileCol + nt * 16 + ln;
            C[(size_t)row * N + col] = acc[nt][r];
        }
    }
}

// ---------------------------------------------------------------------------
// rope + rms_norm: in f32 (T, NHp*64) -> out bf16 (NHp, T, 64)
// One wave per (t, head); lane i handles the rope pair (i, i+32).
// ---------------------------------------------------------------------------
__global__ void __launch_bounds__(32) pp_rope_rms(
    const float* __restrict__ in, const float* __restrict__ cosp,
    const float* __restrict__ sinp, int NHp, unsigned short* __restrict__ outp)
{
    int t  = blockIdx.x;
    int hh = blockIdx.y;
    int i  = threadIdx.x;
    int ld = NHp * HD_;
    float x1 = in[(size_t)t * ld + hh * HD_ + i];
    float x2 = in[(size_t)t * ld + hh * HD_ + i + 32];
    float c  = cosp[t * 32 + i];
    float s  = sinp[t * 32 + i];
    float o1 =  x1 * c + x2 * s;
    float o2 = -x1 * s + x2 * c;
    float ss = o1 * o1 + o2 * o2;
#pragma unroll
    for (int off = 16; off; off >>= 1) ss += __shfl_xor(ss, off, 32);
    float rn = rsqrtf(ss * (1.0f / 64.0f) + EPS_);
    unsigned short* op = outp + ((size_t)hh * T_ + t) * HD_;
    op[i]      = f2bf(o1 * rn);
    op[i + 32] = f2bf(o2 * rn);
}

// ---------------------------------------------------------------------------
// v = v + 2*sigmoid(x[:, :32] @ Wgate) * ve, written TRANSPOSED:
// out bf16 (NKV, HD, T) so P@V B-fragments are contiguous along T.
// ---------------------------------------------------------------------------
__global__ void __launch_bounds__(32) pp_gate_v(
    const float* __restrict__ vf, const float* __restrict__ x,
    const float* __restrict__ ve, const float* __restrict__ Wg,
    unsigned short* __restrict__ vtb)
{
    int t   = blockIdx.x;
    int kvh = blockIdx.y;
    int i   = threadIdx.x;
    float g = x[(size_t)t * NE_ + i] * Wg[i * NKV_ + kvh];
#pragma unroll
    for (int off = 16; off; off >>= 1) g += __shfl_xor(g, off, 32);
    float gate = 2.0f / (1.0f + __expf(-g));
    int base = t * (NKV_ * HD_) + kvh * HD_;
    float v1 = vf[base + i]      + gate * ve[base + i];
    float v2 = vf[base + i + 32] + gate * ve[base + i + 32];
    vtb[((size_t)(kvh * HD_ + i))      * T_ + t] = f2bf(v1);
    vtb[((size_t)(kvh * HD_ + i + 32)) * T_ + t] = f2bf(v2);
}

// ---------------------------------------------------------------------------
// Flash attention (sliding window, causal). Grid (T/64, NH), block = 4 waves.
// Each wave owns 16 query rows of one head; key tiles of 32.
//   S(16x32)  = Q(16x64) @ K^T          -> 4 WMMAs / tile
//   Y(16x64) += P(16x32) @ V(32x64)     -> 4 WMMAs / tile (V pre-transposed)
// K AND V fragment loads are both issued at tile start, so the softmax VALU
// work hides V-load latency (the DS fence would otherwise block hoisting).
// P is reshaped C-layout -> A-layout through a private LDS strip; with a
// 32-halfword stride the A-side re-read is two ds_load_b128 per tile.
// ---------------------------------------------------------------------------
__global__ void __launch_bounds__(128) attn_wmma(
    const unsigned short* __restrict__ Q,   // (NH, T, 64)
    const unsigned short* __restrict__ Km,  // (NKV, T, 64)
    const unsigned short* __restrict__ Vt,  // (NKV, 64, T)
    unsigned short* __restrict__ Y)         // (T, NH*64)
{
    __shared__ __align__(16) unsigned short ldsP[4][16 * 32]; // per-wave strip
    const int lane = threadIdx.x & 31;
    const int w    = threadIdx.x >> 5;
    const int h    = blockIdx.y;
    const int kvh  = h >> 2;                      // NH/NKV = 4
    const int qRow0 = blockIdx.x * 64 + w * 16;
    const int half = (lane >> 4) & 1;
    const int ln   = lane & 15;
    const float NEG = -__builtin_inff();

    // Q fragments for k-steps 0 and 32 (rows are contiguous HD=64 bf16)
    Frag aq0, aq1;
    {
        const unsigned short* qp = Q + ((size_t)(h * T_ + qRow0 + ln)) * HD_;
        int o = half ? 8 : 0;
        aq0.q[0] = *(const uint4*)(qp + o);
        aq0.q[1] = *(const uint4*)(qp + o + 16);
        aq1.q[0] = *(const uint4*)(qp + 32 + o);
        aq1.q[1] = *(const uint4*)(qp + 32 + o + 16);
    }

    v8f accY[4];
#pragma unroll
    for (int i = 0; i < 4; ++i) accY[i] = zero8();
    float rmax[8], rsum[8];
#pragma unroll
    for (int r = 0; r < 8; ++r) { rmax[r] = NEG; rsum[r] = 0.0f; }

    int lo = qRow0 - WIN_;
    if (lo < 0) lo = 0;
    lo &= ~31;

    for (int c0 = lo; c0 <= qRow0 + 15; c0 += 32) {
        // ---- issue ALL K and V fragment loads for this tile up front ------
        Frag bk[2][2];                          // [nt][kstep]
#pragma unroll
        for (int nt = 0; nt < 2; ++nt) {
            const unsigned short* kp =
                Km + ((size_t)(kvh * T_ + c0 + nt * 16 + ln)) * HD_ + half * 8;
            bk[nt][0].q[0] = *(const uint4*)(kp);
            bk[nt][0].q[1] = *(const uint4*)(kp + 16);
            bk[nt][1].q[0] = *(const uint4*)(kp + 32);
            bk[nt][1].q[1] = *(const uint4*)(kp + 48);
        }
        Frag bv[4];
#pragma unroll
        for (int ct = 0; ct < 4; ++ct) {
            const unsigned short* vp =
                Vt + ((size_t)(kvh * HD_ + ct * 16 + ln)) * T_ + c0 + half * 8;
            bv[ct].q[0] = *(const uint4*)(vp);
            bv[ct].q[1] = *(const uint4*)(vp + 16);
        }

        // ---- S = Q @ K^T (scores), two 16-col sub-tiles -------------------
        v8f s[2];
#pragma unroll
        for (int nt = 0; nt < 2; ++nt) {
            v8f z = zero8();
            z = __builtin_amdgcn_wmma_f32_16x16x32_bf16(
                false, aq0.v, false, bk[nt][0].v, (short)0, z, false, false);
            z = __builtin_amdgcn_wmma_f32_16x16x32_bf16(
                false, aq1.v, false, bk[nt][1].v, (short)0, z, false, false);
            s[nt] = z;
        }

        // ---- mask + online softmax ---------------------------------------
        float sv[2][8];
#pragma unroll
        for (int nt = 0; nt < 2; ++nt) {
#pragma unroll
            for (int r = 0; r < 8; ++r) {
                int row = qRow0 + r + half * 8;
                int col = c0 + nt * 16 + ln;
                float v = s[nt][r] * 0.125f;            // HD^-0.5
                bool ok = (col <= row) && (col >= row - WIN_);
                sv[nt][r] = ok ? v : NEG;
            }
        }
#pragma unroll
        for (int r = 0; r < 8; ++r) {
            float tm = fmaxf(sv[0][r], sv[1][r]);
#pragma unroll
            for (int off = 1; off < 16; off <<= 1)
                tm = fmaxf(tm, __shfl_xor(tm, off, 32));
            float nm = fmaxf(rmax[r], tm);
            float corr = (nm == NEG) ? 1.0f : __expf(rmax[r] - nm);
            rmax[r] = nm;
            float ps = 0.0f;
#pragma unroll
            for (int nt = 0; nt < 2; ++nt) {
                float pe = (sv[nt][r] == NEG) ? 0.0f : __expf(sv[nt][r] - nm);
                ps += pe;
                ldsP[w][(r + half * 8) * 32 + nt * 16 + ln] = f2bf(pe);
            }
#pragma unroll
            for (int off = 1; off < 16; off <<= 1)
                ps += __shfl_xor(ps, off, 32);
            rsum[r] = rsum[r] * corr + ps;
#pragma unroll
            for (int ct = 0; ct < 4; ++ct) accY[ct][r] *= corr;
        }

        // per-wave LDS strip: wait for our own DS stores before re-reading
        asm volatile("s_wait_dscnt 0x0" ::: "memory");

        // ---- reload P as A-layout fragment: two ds_load_b128 --------------
        Frag ap;
        {
            const unsigned short* pp = &ldsP[w][ln * 32 + (half ? 8 : 0)];
            ap.q[0] = *(const uint4*)(pp);
            ap.q[1] = *(const uint4*)(pp + 16);
        }

        // ---- Y += P @ V (V loads already in flight since tile start) ------
#pragma unroll
        for (int ct = 0; ct < 4; ++ct) {
            accY[ct] = __builtin_amdgcn_wmma_f32_16x16x32_bf16(
                false, ap.v, false, bv[ct].v, (short)0, accY[ct], false, false);
        }
    }

    // ---- normalize + write y (T, NH*64) bf16 ------------------------------
#pragma unroll
    for (int ct = 0; ct < 4; ++ct) {
#pragma unroll
        for (int r = 0; r < 8; ++r) {
            int row = qRow0 + r + half * 8;
            float val = accY[ct][r] / rsum[r];
            Y[(size_t)row * (NH_ * HD_) + h * HD_ + ct * 16 + ln] = f2bf(val);
        }
    }
}

// ---------------------------------------------------------------------------
// Workspace layout (bytes). Transposed bf16 weights (N x K).
// ---------------------------------------------------------------------------
static const size_t OFF_XB  = 0;                                  // x bf16   4 MB
static const size_t OFF_WQT = OFF_XB  + (size_t)T_ * NE_ * 2;     // WqT bf16 2 MB
static const size_t OFF_WKT = OFF_WQT + (size_t)NE_ * NH_ * HD_ * 2;
static const size_t OFF_WVT = OFF_WKT + (size_t)NE_ * NKV_ * HD_ * 2;
static const size_t OFF_WPT = OFF_WVT + (size_t)NE_ * NKV_ * HD_ * 2;
static const size_t OFF_QF  = OFF_WPT + (size_t)NE_ * NE_ * 2;    // q f32    8 MB
static const size_t OFF_KF  = OFF_QF  + (size_t)T_ * NH_ * HD_ * 4;
static const size_t OFF_VF  = OFF_KF  + (size_t)T_ * NKV_ * HD_ * 4;
static const size_t OFF_QB  = OFF_VF  + (size_t)T_ * NKV_ * HD_ * 4;
static const size_t OFF_KB  = OFF_QB  + (size_t)NH_ * T_ * HD_ * 2;
static const size_t OFF_VTB = OFF_KB  + (size_t)NKV_ * T_ * HD_ * 2;
static const size_t OFF_YB  = OFF_VTB + (size_t)NKV_ * HD_ * T_ * 2;

extern "C" void kernel_launch(void* const* d_in, const int* in_sizes, int n_in,
                              void* d_out, int out_size, void* d_ws, size_t ws_size,
                              hipStream_t stream) {
    const float* x     = (const float*)d_in[0];
    const float* ve    = (const float*)d_in[1];
    const float* cosp  = (const float*)d_in[2];
    const float* sinp  = (const float*)d_in[3];
    const float* Wq    = (const float*)d_in[4];
    const float* Wk    = (const float*)d_in[5];
    const float* Wv    = (const float*)d_in[6];
    const float* Wgate = (const float*)d_in[7];
    const float* Wproj = (const float*)d_in[8];
    float* out = (float*)d_out;

    char* ws = (char*)d_ws;
    unsigned short* xb  = (unsigned short*)(ws + OFF_XB);
    unsigned short* wqt = (unsigned short*)(ws + OFF_WQT);
    unsigned short* wkt = (unsigned short*)(ws + OFF_WKT);
    unsigned short* wvt = (unsigned short*)(ws + OFF_WVT);
    unsigned short* wpt = (unsigned short*)(ws + OFF_WPT);
    float* qf = (float*)(ws + OFF_QF);
    float* kf = (float*)(ws + OFF_KF);
    float* vf = (float*)(ws + OFF_VF);
    unsigned short* qb  = (unsigned short*)(ws + OFF_QB);
    unsigned short* kb  = (unsigned short*)(ws + OFF_KB);
    unsigned short* vtb = (unsigned short*)(ws + OFF_VTB);
    unsigned short* yb  = (unsigned short*)(ws + OFF_YB);

    // 1) x -> bf16; weights -> bf16 transposed (N x K)
    cvt_bf16<<<(T_ * NE_) / 256, 256, 0, stream>>>(x, xb, T_ * NE_);
    {
        dim3 blk(32, 8);
        cvt_bf16_transpose<<<dim3((NH_ * HD_) / 32, NE_ / 32), blk, 0, stream>>>(
            Wq, wqt, NE_, NH_ * HD_);
        cvt_bf16_transpose<<<dim3((NKV_ * HD_) / 32, NE_ / 32), blk, 0, stream>>>(
            Wk, wkt, NE_, NKV_ * HD_);
        cvt_bf16_transpose<<<dim3((NKV_ * HD_) / 32, NE_ / 32), blk, 0, stream>>>(
            Wv, wvt, NE_, NKV_ * HD_);
        cvt_bf16_transpose<<<dim3(NE_ / 32, NE_ / 32), blk, 0, stream>>>(
            Wproj, wpt, NE_, NE_);
    }

    // 2) QKV projections (LDS-free pipelined WMMA GEMM)
    {
        dim3 gq(NH_ * HD_ / 64, T_ / 64);
        gemm_bf16_wmma<<<gq, 128, 0, stream>>>(xb, wqt, qf, T_, NH_ * HD_, NE_);
        dim3 gk(NKV_ * HD_ / 64, T_ / 64);
        gemm_bf16_wmma<<<gk, 128, 0, stream>>>(xb, wkt, kf, T_, NKV_ * HD_, NE_);
        gemm_bf16_wmma<<<gk, 128, 0, stream>>>(xb, wvt, vf, T_, NKV_ * HD_, NE_);
    }

    // 3) rope + rms_norm (q, k), gate + ve (v, transposed)
    pp_rope_rms<<<dim3(T_, NH_), 32, 0, stream>>>(qf, cosp, sinp, NH_, qb);
    pp_rope_rms<<<dim3(T_, NKV_), 32, 0, stream>>>(kf, cosp, sinp, NKV_, kb);
    pp_gate_v<<<dim3(T_, NKV_), 32, 0, stream>>>(vf, x, ve, Wgate, vtb);

    // 4) sliding-window flash attention
    attn_wmma<<<dim3(T_ / 64, NH_), 128, 0, stream>>>(qb, kb, vtb, yb);

    // 5) output projection -> f32 out
    {
        dim3 gp(NE_ / 64, T_ / 64);
        gemm_bf16_wmma<<<gp, 128, 0, stream>>>(yb, wpt, out, T_, NE_, NE_);
    }
}
